// MobileVOSModel_83476984365275
// MI455X (gfx1250) — compile-verified, matrix operand order checked by reading, not run on hardware
//
#include <hip/hip_runtime.h>
#include <hip/hip_bf16.h>
#include <math.h>

// ---------------------------------------------------------------------------
// CDNA5 (gfx1250, wave32) types
// ---------------------------------------------------------------------------
typedef __attribute__((ext_vector_type(16))) __bf16 v16bf;
typedef __attribute__((ext_vector_type(8)))  float  v8f;
typedef unsigned int uint4v __attribute__((ext_vector_type(4)));
typedef unsigned short ushort_t;

__device__ __forceinline__ ushort_t f2bf(float f) {
  unsigned u = __float_as_uint(f);
  u += 0x7FFFu + ((u >> 16) & 1u);   // round-to-nearest-even
  return (ushort_t)(u >> 16);
}

#define BM 128
#define BN 64
#define BK 32
#define LDSTR 40   // 32 K-elems + 8 pad (bf16 units) -> 80B row stride, 16B aligned

// ---------------------------------------------------------------------------
// bf16 GEMM: C(m,n) = sum_k A(m,k)*B(k,n).
// Operands are bf16 in global memory with K-contiguous rows; AOp/BOp return a
// 16B-aligned pointer to an 8-element chunk (OOB conv chunks -> zeroed page,
// branch-free, EXEC stays full).
// Staging: global_load_async_to_lds_b128, double-buffered LDS, software
// pipelined: 6 asyncs/thread/tile, s_wait_asynccnt 6 (in-order completion ->
// tile t landed while tile t+1 is still in flight).
// 128 threads = 4 waves (2x2); wave tile 64x32 = 4x2 v_wmma_f32_16x16x32_bf16
// per k-step (8 WMMA : 12 ds_load_b128).
// ---------------------------------------------------------------------------
template <class AOp, class BOp, class COp>
__global__ __launch_bounds__(128) void gemm_bf16(AOp A, BOp B, COp C,
                                                 int M, int N, int K) {
  __shared__ __align__(16) ushort_t sa[2][BM * LDSTR];
  __shared__ __align__(16) ushort_t sb[2][BN * LDSTR];
  const int tid  = threadIdx.x;
  const int lane = tid & 31;
  const int w    = tid >> 5;
  const int wr   = (w >> 1) * 64;   // wave M offset (2 wave-rows)
  const int wc   = (w & 1) * 32;    // wave N offset (2 wave-cols)
  const int lrow = lane & 15;
  const int lh   = lane >> 4;
  const int bz   = blockIdx.z;
  const int m0   = blockIdx.y * BM;
  const int n0   = blockIdx.x * BN;

  // per-thread chunk coords: A rows {r0,+32,+64,+96}, B rows {r0,+32}
  const int r0  = tid >> 2;
  const int ch0 = (tid & 3) * 8;

  v8f acc[4][2] = {};

  // issue the 6 async chunk loads (4 A + 2 B) for one k-tile
  auto issue = [&](int buf, int k0) {
#pragma unroll
    for (int rr = 0; rr < 4; ++rr) {
      const ushort_t* g = A(bz, m0 + r0 + rr * 32, k0 + ch0);
      unsigned loff = (unsigned)(unsigned long long)(void*)
                      &sa[buf][(r0 + rr * 32) * LDSTR + ch0];
      asm volatile("global_load_async_to_lds_b128 %0, %1, off"
                   :: "v"(loff), "v"(g) : "memory");
    }
#pragma unroll
    for (int rr = 0; rr < 2; ++rr) {
      const ushort_t* g = B(bz, n0 + r0 + rr * 32, k0 + ch0);
      unsigned loff = (unsigned)(unsigned long long)(void*)
                      &sb[buf][(r0 + rr * 32) * LDSTR + ch0];
      asm volatile("global_load_async_to_lds_b128 %0, %1, off"
                   :: "v"(loff), "v"(g) : "memory");
    }
  };

  // fragments (ISA 7.12.2): lane[3:0]=row, lane[4]=K-half; two 16B chunks/lane
  auto compute = [&](int cur) {
    const ushort_t* pa = sa[cur];
    const ushort_t* pb = sb[cur];
    v16bf bfr[2];
#pragma unroll
    for (int nt = 0; nt < 2; ++nt) {
      const ushort_t* ptr = &pb[(wc + nt * 16 + lrow) * LDSTR + lh * 8];
      union { v16bf v; uint4v u[2]; } fr;
      fr.u[0] = *(const uint4v*)(ptr);        // K = lh*8 + 0..7
      fr.u[1] = *(const uint4v*)(ptr + 16);   // K = lh*8 + 16..23
      bfr[nt] = fr.v;
    }
#pragma unroll
    for (int mt = 0; mt < 4; ++mt) {
      const ushort_t* ptr = &pa[(wr + mt * 16 + lrow) * LDSTR + lh * 8];
      union { v16bf v; uint4v u[2]; } fr;
      fr.u[0] = *(const uint4v*)(ptr);
      fr.u[1] = *(const uint4v*)(ptr + 16);
      v16bf af = fr.v;
#pragma unroll
      for (int nt = 0; nt < 2; ++nt)
        acc[mt][nt] = __builtin_amdgcn_wmma_f32_16x16x32_bf16(
            false, af, false, bfr[nt], (short)0, acc[mt][nt], false, false);
    }
  };

  const int NT = K / BK;   // >= 2 for every GEMM in this model
  issue(0, 0);
  for (int t = 0; t < NT - 1; ++t) {
    const int cur = t & 1;
    issue(cur ^ 1, (t + 1) * BK);                       // prefetch next tile
    asm volatile("s_wait_asynccnt 0x6" ::: "memory");   // tile t landed
    __syncthreads();
    compute(cur);
    __syncthreads();   // protect buffer before re-issue next iteration
  }
  asm volatile("s_wait_asynccnt 0x0" ::: "memory");
  __syncthreads();
  compute((NT - 1) & 1);

  // D layout: vgpr i -> M = i + lh*8 ; lane[3:0] -> N  (M,N multiples of tile)
#pragma unroll
  for (int mt = 0; mt < 4; ++mt)
#pragma unroll
    for (int nt = 0; nt < 2; ++nt)
#pragma unroll
      for (int i = 0; i < 8; ++i) {
        int m = m0 + wr + mt * 16 + lh * 8 + i;
        int n = n0 + wc + nt * 16 + lrow;
        C(bz, m, n, acc[mt][nt][i]);
      }
}

// ---------------------------------------------------------------------------
// Operand address functors (all bf16, return 16B-aligned 8-elem chunk)
// ---------------------------------------------------------------------------

// generic row-major [row][ldk] with row offset
struct ARow {
  const ushort_t* a; int ldk; int roff;
  __device__ const ushort_t* operator()(int, int m, int k) const {
    return a + (roff + m) * ldk + k;
  }
};
struct BRow {
  const ushort_t* w; int ldk; int roff;
  __device__ const ushort_t* operator()(int, int n, int k) const {
    return w + (roff + n) * ldk + k;
  }
};

// patch-embed A: bf16 frame [8,3,512,512]; m=(b,oy,ox), k=(c,dy,dx)
struct AImgB {
  const ushort_t* img;
  __device__ const ushort_t* operator()(int, int m, int k) const {
    int b = m >> 10, pp = m & 1023, oy = pp >> 5, ox = pp & 31;
    int c = k >> 8, r = k & 255, dy = r >> 4, dx = r & 15;
    return img + ((b * 3 + c) * 512 + oy * 16 + dy) * 512 + ox * 16 + dx;
  }
};

// 3x3 SAME conv A, channel-last bf16 [b*1024 + y*32 + x][Cin], k = t*Cin + c.
// Out-of-bounds -> zeroed scratch page (branch-free staging).
struct AConv3B {
  const ushort_t* in; const ushort_t* zpad; int cshift;
  __device__ const ushort_t* operator()(int, int m, int k) const {
    int b = m >> 10, pp = m & 1023, oy = pp >> 5, ox = pp & 31;
    int t = k >> cshift, c = k & ((1 << cshift) - 1);
    int dy = t / 3, dx = t - dy * 3;
    int y = oy + dy - 1, x = ox + dx - 1;
    if ((unsigned)y > 31u || (unsigned)x > 31u) return zpad;
    return in + (b * 1024 + y * 32 + x) * (1 << cshift) + c;
  }
};

// ---------------------------------------------------------------------------
// Epilogues
// ---------------------------------------------------------------------------

// encoder: bias+ReLU -> featT bf16 [m][512]
struct CFeatT {
  const float* bias; ushort_t* out;
  __device__ void operator()(int, int m, int n, float v) const {
    v += bias[n]; v = v > 0.f ? v : 0.f;
    out[m * 512 + n] = f2bf(v);
  }
};

// key/val for memory frames: key f32 [b,64,2048] (pre-norm), val bf16 [b,256,2048]
struct CKVMem {
  const float* kb; const float* vb; const float* mask;
  float* keyo; int koff; ushort_t* valo; int voff;
  __device__ void operator()(int, int m, int n, float v) const {
    int b = m >> 10, pp = m & 1023, oy = pp >> 5, ox = pp & 31;
    float mv = mask[(b * 512 + oy * 16) * 512 + ox * 16];
    if (n < 64) {
      keyo[(b * 64 + n) * 2048 + koff + pp] = (v + kb[n]) * mv;
    } else {
      int c = n - 64;
      valo[(b * 256 + c) * 2048 + voff + pp] = f2bf((v + vb[c]) * mv);
    }
  }
};

// key/val for query frame: key f32 [b,64,1024], qv -> fused channel-last bf16
struct CKVQuery {
  const float* kb; const float* vb;
  float* keyo; ushort_t* fused;
  __device__ void operator()(int, int m, int n, float v) const {
    int b = m >> 10, pp = m & 1023;
    if (n < 64) {
      keyo[(b * 64 + n) * 1024 + pp] = v + kb[n];
    } else {
      fused[m * 512 + (n - 64)] = f2bf(v + vb[n - 64]);
    }
  }
};

// affinity: scale 1/sqrt(64), f32 (pre-softmax)
struct CAttn {
  float* attn;
  __device__ void operator()(int, int m, int n, float v) const {
    attn[m * 2048 + n] = v * 0.125f;
  }
};

// mem_read -> fused channel-last bf16, channels 256..511
struct CFuse {
  ushort_t* fused; int bb;
  __device__ void operator()(int, int m, int n, float v) const {
    fused[(bb * 1024 + m) * 512 + 256 + n] = f2bf(v);
  }
};

// dec1: bias+ReLU -> dbuf channel-last bf16 [m][256]
struct CDbuf {
  const float* bias; ushort_t* out;
  __device__ void operator()(int, int m, int n, float v) const {
    v += bias[n]; v = v > 0.f ? v : 0.f;
    out[m * 256 + n] = f2bf(v);
  }
};

// dec2: bias+ReLU -> decf f32 channel-major [b,256,32,32]
struct CDecF {
  const float* bias; float* out;
  __device__ void operator()(int, int m, int n, float v) const {
    int b = m >> 10, pp = m & 1023;
    v += bias[n]; v = v > 0.f ? v : 0.f;
    out[(b * 256 + n) * 1024 + pp] = v;
  }
};

// ---------------------------------------------------------------------------
// Prep / pointwise kernels
// ---------------------------------------------------------------------------
__global__ void zero_fill32(ushort_t* p) {
  if (threadIdx.x < 32) p[threadIdx.x] = 0;
}

__global__ __launch_bounds__(256) void f32_to_bf16_k(const float* s, ushort_t* d, int n) {
  int i = blockIdx.x * 256 + threadIdx.x;
  if (i < n) d[i] = f2bf(s[i]);
}

// combined key|val weights -> [320][512] bf16
__global__ __launch_bounds__(256) void prep_kv(const float* kw, const float* vw, ushort_t* d) {
  int i = blockIdx.x * 256 + threadIdx.x;
  if (i >= 320 * 512) return;
  int n = i >> 9, k = i & 511;
  d[i] = f2bf(n < 64 ? kw[n * 512 + k] : vw[(n - 64) * 512 + k]);
}

// decoder weights (n, c, dy, dx) -> [n][t*Cin + c] bf16
__global__ __launch_bounds__(256) void prep_dec(const float* w, ushort_t* d, int Cin) {
  int total = 256 * 9 * Cin;
  int i = blockIdx.x * 256 + threadIdx.x;
  if (i >= total) return;
  int n = i / (9 * Cin);
  int rem = i - n * 9 * Cin;
  int t = rem / Cin, c = rem - t * Cin;
  d[i] = f2bf(w[(n * Cin + c) * 9 + t]);
}

// L2-normalize each (b,:,m) column of f32 src [b][C][ld]; write bf16 transposed [b*ld+m][C]
__global__ __launch_bounds__(256) void l2norm_T(const float* src, ushort_t* dst,
                                                int C, int ld, int total) {
  int j = blockIdx.x * 256 + threadIdx.x;
  if (j >= total) return;
  int b = j / ld, m = j - b * ld;
  const float* basep = src + (b * C) * ld + m;
  float s = 0.f;
  for (int c = 0; c < C; ++c) { float v = basep[c * ld]; s += v * v; }
  float inv = 1.f / fmaxf(sqrtf(s), 1e-12f);
  ushort_t* drow = dst + j * C;
  for (int c = 0; c < C; ++c) drow[c] = f2bf(basep[c * ld] * inv);
}

// softmax over rows of 2048; also emit bf16 copy for the next GEMM's A operand
__global__ __launch_bounds__(256) void softmax_rows(float* attn, ushort_t* attnb, int ncols) {
  float* p = attn + blockIdx.x * ncols;
  ushort_t* ob = attnb + blockIdx.x * ncols;
  __shared__ float red[256];
  int tid = threadIdx.x;
  float mx = -1e30f;
  for (int j = tid; j < ncols; j += 256) mx = fmaxf(mx, p[j]);
  red[tid] = mx; __syncthreads();
  for (int s = 128; s > 0; s >>= 1) { if (tid < s) red[tid] = fmaxf(red[tid], red[tid + s]); __syncthreads(); }
  mx = red[0]; __syncthreads();
  float sum = 0.f;
  for (int j = tid; j < ncols; j += 256) { float e = __expf(p[j] - mx); p[j] = e; sum += e; }
  red[tid] = sum; __syncthreads();
  for (int s = 128; s > 0; s >>= 1) { if (tid < s) red[tid] += red[tid + s]; __syncthreads(); }
  float inv = 1.f / red[0];
  for (int j = tid; j < ncols; j += 256) ob[j] = f2bf(p[j] * inv);
}

// 1x1 conv 256->2 (no relu)
__global__ __launch_bounds__(256) void cls_head(const float* feat, const float* w,
                                                const float* bias, float* logits) {
  int i = blockIdx.x * 256 + threadIdx.x;
  if (i >= 8192) return;
  int b = i >> 10, pp = i & 1023;
  const float* f = feat + (b * 256) * 1024 + pp;
  for (int o = 0; o < 2; ++o) {
    float s = bias[o];
    const float* wr = w + o * 256;
    for (int c = 0; c < 256; ++c) s += f[c * 1024] * wr[c];
    logits[(b * 2 + o) * 1024 + pp] = s;
  }
}

__global__ __launch_bounds__(256) void rep_l2norm(const float* feat, float* out) {
  int i = blockIdx.x * 256 + threadIdx.x;
  if (i >= 8192) return;
  int b = i >> 10, pp = i & 1023;
  const float* f = feat + (b * 256) * 1024 + pp;
  float s = 0.f;
  for (int c = 0; c < 256; ++c) { float v = f[c * 1024]; s += v * v; }
  float inv = 1.f / fmaxf(sqrtf(s), 1e-12f);
  for (int c = 0; c < 256; ++c) out[(b * 256 + c) * 1024 + pp] = f[c * 1024] * inv;
}

// bilinear 32x32 -> 512x512 (half-pixel centers, clamp-to-edge)
__global__ __launch_bounds__(256) void upsample_bilinear(const float* logits, float* out) {
  int i = blockIdx.x * 256 + threadIdx.x;
  if (i >= 8 * 2 * 512 * 512) return;
  int X = i & 511, t = i >> 9, Y = t & 511, bo = t >> 9;
  float sx = fminf(fmaxf((X + 0.5f) * (1.f / 16.f) - 0.5f, 0.f), 31.f);
  float sy = fminf(fmaxf((Y + 0.5f) * (1.f / 16.f) - 0.5f, 0.f), 31.f);
  int x0 = (int)sx, y0 = (int)sy;
  int x1 = x0 + 1 > 31 ? 31 : x0 + 1;
  int y1 = y0 + 1 > 31 ? 31 : y0 + 1;
  float wx = sx - x0, wy = sy - y0;
  const float* src = logits + bo * 1024;
  float v0 = src[y0 * 32 + x0] * (1.f - wx) + src[y0 * 32 + x1] * wx;
  float v1 = src[y1 * 32 + x0] * (1.f - wx) + src[y1 * 32 + x1] * wx;
  out[i] = v0 * (1.f - wy) + v1 * wy;
}

// ---------------------------------------------------------------------------
// Launcher
// ---------------------------------------------------------------------------
extern "C" void kernel_launch(void* const* d_in, const int* in_sizes, int n_in,
                              void* d_out, int out_size, void* d_ws, size_t ws_size,
                              hipStream_t stream) {
  (void)in_sizes; (void)n_in; (void)out_size; (void)ws_size;
  const float* query  = (const float*)d_in[0];
  const float* prevf  = (const float*)d_in[1];
  const float* prevm  = (const float*)d_in[2];
  const float* firstf = (const float*)d_in[3];
  const float* firstm = (const float*)d_in[4];
  const float* eqw = (const float*)d_in[5];
  const float* eqb = (const float*)d_in[6];
  const float* emw = (const float*)d_in[7];
  const float* emb = (const float*)d_in[8];
  const float* kw  = (const float*)d_in[9];
  const float* kb  = (const float*)d_in[10];
  const float* vw  = (const float*)d_in[11];
  const float* vb  = (const float*)d_in[12];
  const float* d1w = (const float*)d_in[13];
  const float* d1b = (const float*)d_in[14];
  const float* d2w = (const float*)d_in[15];
  const float* d2b = (const float*)d_in[16];
  const float* cw  = (const float*)d_in[17];
  const float* cb  = (const float*)d_in[18];

  // bump allocator over workspace (256B aligned)
  char* wbase = (char*)d_ws;
  size_t woff = 0;
  auto walloc = [&](size_t bytes) {
    void* r = wbase + woff;
    woff += (bytes + 255) & ~(size_t)255;
    return r;
  };

  ushort_t* zpad      = (ushort_t*)walloc(64);                        // zeroed pad page
  ushort_t* frame_bf  = (ushort_t*)walloc(8UL * 3 * 512 * 512 * 2);   // per-frame bf16 (reused)
  ushort_t* ewq_bf    = (ushort_t*)walloc(512UL * 768 * 2);
  ushort_t* ewm_bf    = (ushort_t*)walloc(512UL * 768 * 2);
  ushort_t* kvw_bf    = (ushort_t*)walloc(320UL * 512 * 2);
  ushort_t* d1w_bf    = (ushort_t*)walloc(256UL * 9 * 512 * 2);
  ushort_t* d2w_bf    = (ushort_t*)walloc(256UL * 9 * 256 * 2);
  ushort_t* featT     = (ushort_t*)walloc(8192UL * 512 * 2);          // [m][512]
  float*    memkey    = (float*)   walloc(8UL * 64 * 2048 * 4);       // pre-norm f32
  float*    qkey      = (float*)   walloc(8UL * 64 * 1024 * 4);
  ushort_t* memkeyT   = (ushort_t*)walloc(8UL * 2048 * 64 * 2);       // normalized, [b*2048+m][64]
  ushort_t* qkeyT     = (ushort_t*)walloc(8UL * 1024 * 64 * 2);       // normalized, [b*1024+q][64]
  ushort_t* memval_bf = (ushort_t*)walloc(8UL * 256 * 2048 * 2);      // [b,256,2048]
  ushort_t* fused_cl  = (ushort_t*)walloc(8192UL * 512 * 2);          // channel-last
  float*    attn      = (float*)   walloc(1024UL * 2048 * 4);         // per-batch
  ushort_t* attn_bf   = (ushort_t*)walloc(1024UL * 2048 * 2);
  ushort_t* dbuf_cl   = (ushort_t*)walloc(8192UL * 256 * 2);
  float*    decf      = (float*)   walloc(8UL * 256 * 1024 * 4);
  float*    logit     = (float*)   walloc(8UL * 2 * 1024 * 4);

  float* rep_out = (float*)d_out;                    // [8,256,32,32]
  float* log_out = (float*)d_out + 8 * 256 * 1024;   // [8,2,512,512]

  dim3 blk(128);

  // ---- prep: zero pad page + bf16 weights in GEMM-friendly layouts ----
  zero_fill32<<<dim3(1), dim3(32), 0, stream>>>(zpad);
  f32_to_bf16_k<<<dim3((512 * 768 + 255) / 256), dim3(256), 0, stream>>>(eqw, ewq_bf, 512 * 768);
  f32_to_bf16_k<<<dim3((512 * 768 + 255) / 256), dim3(256), 0, stream>>>(emw, ewm_bf, 512 * 768);
  prep_kv<<<dim3((320 * 512 + 255) / 256), dim3(256), 0, stream>>>(kw, vw, kvw_bf);
  prep_dec<<<dim3((256 * 9 * 512 + 255) / 256), dim3(256), 0, stream>>>(d1w, d1w_bf, 512);
  prep_dec<<<dim3((256 * 9 * 256 + 255) / 256), dim3(256), 0, stream>>>(d2w, d2w_bf, 256);

  // ---- encode three frames ----
  const int FN = 8 * 3 * 512 * 512;
  struct Enc { const float* frame; const ushort_t* ew; const float* eb; };
  Enc encs[3] = { { firstf, ewm_bf, emb }, { prevf, ewm_bf, emb }, { query, ewq_bf, eqb } };
  for (int f = 0; f < 3; ++f) {
    f32_to_bf16_k<<<dim3((FN + 255) / 256), dim3(256), 0, stream>>>(encs[f].frame, frame_bf, FN);
    // patch-embed implicit GEMM 8192x512x768
    gemm_bf16<<<dim3(512 / BN, 8192 / BM, 1), blk, 0, stream>>>(
        AImgB{frame_bf}, BRow{encs[f].ew, 768, 0}, CFeatT{encs[f].eb, featT},
        8192, 512, 768);
    // key|val 1x1 GEMM 8192x320x512
    if (f == 0)
      gemm_bf16<<<dim3(320 / BN, 8192 / BM, 1), blk, 0, stream>>>(
          ARow{featT, 512, 0}, BRow{kvw_bf, 512, 0},
          CKVMem{kb, vb, firstm, memkey, 0, memval_bf, 0}, 8192, 320, 512);
    else if (f == 1)
      gemm_bf16<<<dim3(320 / BN, 8192 / BM, 1), blk, 0, stream>>>(
          ARow{featT, 512, 0}, BRow{kvw_bf, 512, 0},
          CKVMem{kb, vb, prevm, memkey, 1024, memval_bf, 1024}, 8192, 320, 512);
    else
      gemm_bf16<<<dim3(320 / BN, 8192 / BM, 1), blk, 0, stream>>>(
          ARow{featT, 512, 0}, BRow{kvw_bf, 512, 0},
          CKVQuery{kb, vb, qkey, fused_cl}, 8192, 320, 512);
  }

  // ---- key L2-normalize + transpose to bf16 ----
  l2norm_T<<<dim3((8 * 2048 + 255) / 256), dim3(256), 0, stream>>>(memkey, memkeyT, 64, 2048, 8 * 2048);
  l2norm_T<<<dim3((8 * 1024 + 255) / 256), dim3(256), 0, stream>>>(qkey, qkeyT, 64, 1024, 8 * 1024);

  // ---- attention per batch (reuses attn buffers) ----
  for (int b = 0; b < 8; ++b) {
    gemm_bf16<<<dim3(2048 / BN, 1024 / BM, 1), blk, 0, stream>>>(
        ARow{qkeyT, 64, b * 1024}, BRow{memkeyT, 64, b * 2048}, CAttn{attn},
        1024, 2048, 64);
    softmax_rows<<<dim3(1024), dim3(256), 0, stream>>>(attn, attn_bf, 2048);
    gemm_bf16<<<dim3(256 / BN, 1024 / BM, 1), blk, 0, stream>>>(
        ARow{attn_bf, 2048, 0}, BRow{memval_bf, 2048, b * 256}, CFuse{fused_cl, b},
        1024, 256, 2048);
  }

  // ---- decoder 3x3 convs (implicit GEMM, channel-last bf16 activations) ----
  gemm_bf16<<<dim3(256 / BN, 8192 / BM, 1), blk, 0, stream>>>(
      AConv3B{fused_cl, zpad, 9}, BRow{d1w_bf, 9 * 512, 0}, CDbuf{d1b, dbuf_cl},
      8192, 256, 9 * 512);
  gemm_bf16<<<dim3(256 / BN, 8192 / BM, 1), blk, 0, stream>>>(
      AConv3B{dbuf_cl, zpad, 8}, BRow{d2w_bf, 9 * 256, 0}, CDecF{d2b, decf},
      8192, 256, 9 * 256);

  // ---- heads ----
  cls_head<<<dim3(32), dim3(256), 0, stream>>>(decf, cw, cb, logit);
  rep_l2norm<<<dim3(32), dim3(256), 0, stream>>>(decf, rep_out);
  upsample_bilinear<<<dim3(8 * 2 * 512 * 512 / 256), dim3(256), 0, stream>>>(logit, log_out);
}
